// LATTE_51685636440227
// MI455X (gfx1250) — compile-verified, matrix operand order checked by reading, not run on hardware
//
#include <hip/hip_runtime.h>
#include <hip/hip_bf16.h>

#define NNODES 100000
#define EEDGES 800000
#define DDIM   256
#define HEADS  4
#define CHAN   64   // DDIM / HEADS

typedef __attribute__((ext_vector_type(16))) __bf16        v16bf;
typedef __attribute__((ext_vector_type(8)))  float         v8f;
typedef __attribute__((ext_vector_type(4)))  unsigned int  u32x4;
typedef __attribute__((ext_vector_type(8)))  int           i32x8;
typedef __attribute__((ext_vector_type(4)))  int           i32x4;

// ---------- helpers ----------
__device__ __forceinline__ float lrelu(float x) { return x >= 0.f ? x : 0.2f * x; }

__device__ __forceinline__ void atomicMaxF(float* p, float v) {
    if (v >= 0.f) atomicMax((int*)p, __float_as_int(v));
    else          atomicMin((unsigned int*)p, __float_as_uint(v));
}
__device__ __forceinline__ void atomicAddF(float* p, float v) {
    __hip_atomic_fetch_add(p, v, __ATOMIC_RELAXED, __HIP_MEMORY_SCOPE_AGENT);
}
__device__ __forceinline__ float red8(float v) {  // sum across 8-lane head group
    v += __shfl_xor(v, 1);
    v += __shfl_xor(v, 2);
    v += __shfl_xor(v, 4);
    return v;
}
__device__ __forceinline__ unsigned pack_bf16(float lo, float hi) {
    union { __bf16 b[2]; unsigned u; } c;
    c.b[0] = (__bf16)lo;
    c.b[1] = (__bf16)hi;
    return c.u;
}

// ---------- TDM: 2D tile load (bf16 elements) Global -> LDS ----------
// tile_d0 elements along contiguous dim0 (stride0 element pitch between rows),
// tile_d1 rows. tensor_d0/d1 bound the OOB check (reads past extent -> 0).
__device__ __forceinline__ void tdm_load_2d_bf16(
    const void* gaddr, unsigned lds_off,
    unsigned tensor_d0, unsigned tensor_d1, unsigned stride0,
    unsigned tile_d0, unsigned tile_d1)
{
    unsigned long long ga = (unsigned long long)(size_t)gaddr;
    u32x4 g0;
    g0[0] = 1u;                                            // count=1, user descriptor
    g0[1] = lds_off;                                       // lds_addr (bytes)
    g0[2] = (unsigned)(ga & 0xFFFFFFFFu);                  // global_addr[31:0]
    g0[3] = (unsigned)((ga >> 32) & 0x01FFFFFFu) | (2u << 30); // addr[56:32] | type=2
    i32x8 g1;
    g1[0] = (int)(1u << 16);                               // wg_mask=0, data_size=1 (2B)
    g1[1] = (int)((tensor_d0 & 0xFFFFu) << 16);            // abar=0 | tensor_dim0[15:0]
    g1[2] = (int)((tensor_d0 >> 16) | ((tensor_d1 & 0xFFFFu) << 16));
    g1[3] = (int)((tensor_d1 >> 16) | (tile_d0 << 16));    // tensor_dim1 hi | tile_dim0
    g1[4] = (int)(tile_d1 & 0xFFFFu);                      // tile_dim1 | tile_dim2=0
    g1[5] = (int)stride0;                                  // tensor_dim0_stride[31:0]
    g1[6] = 0;                                             // stride0 hi | dim1_stride lo
    g1[7] = 0;
    i32x4 z4 = {0, 0, 0, 0};
    i32x8 z8 = {0, 0, 0, 0, 0, 0, 0, 0};
    __builtin_amdgcn_tensor_load_to_lds(g0, g1, z4, z4, z8, 0);
}

// ---------- kernel 0: streaming fp32 -> bf16 conversion ----------
__global__ __launch_bounds__(256)
void f32_to_bf16_kernel(const float* __restrict__ src,
                        unsigned short* __restrict__ dst, int n4)
{
    const int i = blockIdx.x * 256 + threadIdx.x;
    if (i >= n4) return;
    float4 v = ((const float4*)src)[i];
    ((uint2*)dst)[i] = make_uint2(pack_bf16(v.x, v.y), pack_bf16(v.z, v.w));
}

// ---------- kernel 1: dual projection GEMM, TDM-staged, bf16 WMMA ----------
// out[row, col] = sum_k x[row,k] * W[col,k] + b[col]   (z=0 -> l/W_l, z=1 -> r/W_r)
__global__ __launch_bounds__(128)
void proj_gemm_tdm_kernel(const unsigned short* __restrict__ xbf,
                          const unsigned short* __restrict__ wbf,
                          const float* __restrict__ bl, const float* __restrict__ br,
                          float* __restrict__ lout, float* __restrict__ rout)
{
    const unsigned short* Wsel = wbf + (size_t)blockIdx.z * DDIM * DDIM;
    const float* bias = (blockIdx.z == 0) ? bl : br;
    float*       out  = (blockIdx.z == 0) ? lout : rout;

    __shared__ __align__(16) unsigned short As[2][64 * 32];  // [m][k] bf16
    __shared__ __align__(16) unsigned short Bs[2][64 * 32];  // [col][k] bf16

    const int tid     = threadIdx.x;
    const int lane    = tid & 31;
    const int wave    = tid >> 5;          // 0..3
    const int rowBase = blockIdx.x * 64;
    const int colBase = blockIdx.y * 64;
    const unsigned rowsA = (unsigned)((NNODES - rowBase) < 64 ? (NNODES - rowBase) : 64);

    v8f acc[4] = {};

    // double-buffered TDM pipeline: wave 0 drives the DMA, all waves compute
    if (wave == 0) {
        tdm_load_2d_bf16(xbf + (size_t)rowBase * DDIM,
                         (unsigned)(size_t)&As[0][0], 32, rowsA, DDIM, 32, 64);
        tdm_load_2d_bf16(Wsel + (size_t)colBase * DDIM,
                         (unsigned)(size_t)&Bs[0][0], 32, 64, DDIM, 32, 64);
    }

    for (int it = 0; it < DDIM / 32; ++it) {
        const int cur = it & 1;
        if (wave == 0) __builtin_amdgcn_s_wait_tensorcnt(0);
        __syncthreads();                      // LDS tile visible to all waves
        if (wave == 0 && it + 1 < DDIM / 32) {
            const int k1 = (it + 1) * 32;
            tdm_load_2d_bf16(xbf + (size_t)rowBase * DDIM + k1,
                             (unsigned)(size_t)&As[1 - cur][0], 32, rowsA, DDIM, 32, 64);
            tdm_load_2d_bf16(Wsel + (size_t)colBase * DDIM + k1,
                             (unsigned)(size_t)&Bs[1 - cur][0], 32, 64, DDIM, 32, 64);
        }

        // ---- WMMA: each wave computes a 16x64 strip ----
        const int m  = wave * 16 + (lane & 15);
        const int kb = (lane < 16) ? 0 : 8;   // K split per ISA 16-bit A/B layout
        union { uint4 u[2]; v16bf v; } afrag;
        afrag.u[0] = *(const uint4*)(&As[cur][m * 32 + kb]);
        afrag.u[1] = *(const uint4*)(&As[cur][m * 32 + kb + 16]);
        #pragma unroll
        for (int t = 0; t < 4; ++t) {
            const int c = t * 16 + (lane & 15);
            union { uint4 u[2]; v16bf v; } bfrag;
            bfrag.u[0] = *(const uint4*)(&Bs[cur][c * 32 + kb]);
            bfrag.u[1] = *(const uint4*)(&Bs[cur][c * 32 + kb + 16]);
            acc[t] = __builtin_amdgcn_wmma_f32_16x16x32_bf16(
                false, afrag.v, false, bfrag.v, (short)0, acc[t], false, false);
        }
        __syncthreads();                      // compute done before buffer reuse
    }

    // ---- epilogue: C layout (VGPR i -> M=i lanes 0-15, M=8+i lanes 16-31) ----
    #pragma unroll
    for (int t = 0; t < 4; ++t) {
        const int col = colBase + t * 16 + (lane & 15);
        const float bv = bias[col];
        #pragma unroll
        for (int i = 0; i < 8; ++i) {
            const int row = rowBase + wave * 16 + ((lane < 16) ? i : (8 + i));
            if (row < NNODES) out[(size_t)row * DDIM + col] = acc[t][i] + bv;
        }
    }
}

// ---------- kernel 2a: per-edge logits + segment max (one wave per edge) ----------
__global__ __launch_bounds__(256)
void edge_logits_kernel(const int* __restrict__ ei, const float* __restrict__ l,
                        const float* __restrict__ r, const float* __restrict__ attn_m,
                        float* __restrict__ logits, float* __restrict__ mx)
{
    const int e = blockIdx.x * 8 + (threadIdx.x >> 5);
    if (e >= EEDGES) return;
    const int lane = threadIdx.x & 31;
    const int src  = ei[e];
    const int tgt  = ei[EEDGES + e];
    const int head = lane >> 3;
    const int c8   = lane * 8;
    const int cc   = (lane & 7) * 8;

    const float4* rv = (const float4*)(r + (size_t)tgt * DDIM + c8);
    const float4* lv = (const float4*)(l + (size_t)src * DDIM + c8);
    float4 r0 = rv[0], r1 = rv[1];
    float4 l0 = lv[0], l1 = lv[1];
    float rr[8] = {r0.x, r0.y, r0.z, r0.w, r1.x, r1.y, r1.z, r1.w};
    float ll[8] = {l0.x, l0.y, l0.z, l0.w, l1.x, l1.y, l1.z, l1.w};

    const float* ai_w = attn_m + head * 2 * CHAN + cc;         // attn[m,h,:C]
    const float* aj_w = attn_m + head * 2 * CHAN + CHAN + cc;  // attn[m,h,C:]
    float s = 0.f;
    #pragma unroll
    for (int j = 0; j < 8; ++j)
        s += lrelu(rr[j]) * ai_w[j] + lrelu(ll[j]) * aj_w[j];
    s = red8(s);
    if ((lane & 7) == 0) {
        logits[(size_t)e * HEADS + head] = s;
        atomicMaxF(mx + (size_t)tgt * HEADS + head, s);
    }
}

// ---------- kernel 2b: exp(logit - max) + denominator ----------
__global__ __launch_bounds__(256)
void edge_exp_kernel(const int* __restrict__ ei, float* __restrict__ exbuf,
                     const float* __restrict__ mx, float* __restrict__ den)
{
    const int t = blockIdx.x * 256 + threadIdx.x;   // E*H = 3.2M
    if (t >= EEDGES * HEADS) return;
    const int e = t >> 2;
    const int h = t & 3;
    const int tgt = ei[EEDGES + e];
    float m = mx[(size_t)tgt * HEADS + h];
    if (!(m > -1e30f)) m = 0.f;                     // empty-segment guard
    const float ex = __expf(exbuf[t] - m);
    exbuf[t] = ex;
    atomicAddF(den + (size_t)tgt * HEADS + h, ex);
}

// ---------- kernel 2c: alpha-weighted message scatter (one wave per edge) ----------
__global__ __launch_bounds__(256)
void edge_scatter_kernel(const int* __restrict__ ei, const float* __restrict__ l,
                         const float* __restrict__ exbuf, const float* __restrict__ den,
                         float* __restrict__ outm)
{
    const int e = blockIdx.x * 8 + (threadIdx.x >> 5);
    if (e >= EEDGES) return;
    const int lane = threadIdx.x & 31;
    const int src  = ei[e];
    const int tgt  = ei[EEDGES + e];
    const int head = lane >> 3;
    const int c8   = lane * 8;

    const float alpha = exbuf[(size_t)e * HEADS + head] /
                        (den[(size_t)tgt * HEADS + head] + 1e-16f);
    const float4* lv = (const float4*)(l + (size_t)src * DDIM + c8);
    float4 l0 = lv[0], l1 = lv[1];
    float ll[8] = {l0.x, l0.y, l0.z, l0.w, l1.x, l1.y, l1.z, l1.w};
    float* op = outm + (size_t)tgt * DDIM + c8;
    #pragma unroll
    for (int j = 0; j < 8; ++j) atomicAddF(op + j, ll[j] * alpha);
}

// ---------- kernel 3: relation-level beta softmax + combine + ReLU ----------
__global__ __launch_bounds__(256)
void final_kernel(const float* __restrict__ r, const float* __restrict__ o0,
                  const float* __restrict__ o1, const float* __restrict__ ral,
                  const float* __restrict__ rar, float* __restrict__ out)
{
    const int n = blockIdx.x * 8 + (threadIdx.x >> 5);
    if (n >= NNODES) return;
    const int lane = threadIdx.x & 31;
    const int head = lane >> 3;
    const int c8   = lane * 8;
    const int cc   = (lane & 7) * 8;

    const float4* rv = (const float4*)(r  + (size_t)n * DDIM + c8);
    const float4* v0 = (const float4*)(o0 + (size_t)n * DDIM + c8);
    const float4* v1 = (const float4*)(o1 + (size_t)n * DDIM + c8);
    float4 ra = rv[0], rb = rv[1];
    float4 aa = v0[0], ab = v0[1];
    float4 ba = v1[0], bb = v1[1];
    float rr[8] = {ra.x, ra.y, ra.z, ra.w, rb.x, rb.y, rb.z, rb.w};
    float a0[8] = {aa.x, aa.y, aa.z, aa.w, ab.x, ab.y, ab.z, ab.w};
    float a1[8] = {ba.x, ba.y, ba.z, ba.w, bb.x, bb.y, bb.z, bb.w};

    const float* wl = ral + head * CHAN + cc;
    const float* wr = rar + head * CHAN + cc;
    float al = 0.f, q0 = 0.f, q1 = 0.f, q2 = 0.f;
    #pragma unroll
    for (int j = 0; j < 8; ++j) {
        al += rr[j] * wl[j];
        q0 += a0[j] * wr[j];
        q1 += a1[j] * wr[j];
        q2 += rr[j] * wr[j];
    }
    al = red8(al); q0 = red8(q0); q1 = red8(q1); q2 = red8(q2);

    const float s0 = lrelu(al + q0), s1 = lrelu(al + q1), s2 = lrelu(al + q2);
    const float mm = fmaxf(s0, fmaxf(s1, s2));
    const float e0 = __expf(s0 - mm), e1 = __expf(s1 - mm), e2 = __expf(s2 - mm);
    const float inv = 1.f / (e0 + e1 + e2);
    const float b0 = e0 * inv, b1 = e1 * inv, b2 = e2 * inv;

    float res[8];
    #pragma unroll
    for (int j = 0; j < 8; ++j) {
        float v = a0[j] * b0 + a1[j] * b1 + rr[j] * b2;
        res[j] = v > 0.f ? v : 0.f;
    }
    float4* op = (float4*)(out + (size_t)n * DDIM + c8);
    op[0] = make_float4(res[0], res[1], res[2], res[3]);
    op[1] = make_float4(res[4], res[5], res[6], res[7]);
}

// ---------- -inf fill (for segment max init) ----------
__global__ __launch_bounds__(256)
void fill_neginf_kernel(unsigned int* __restrict__ p, int n)
{
    const int i = blockIdx.x * 256 + threadIdx.x;
    if (i < n) p[i] = 0xFF800000u;   // -inf
}

extern "C" void kernel_launch(void* const* d_in, const int* in_sizes, int n_in,
                              void* d_out, int out_size, void* d_ws, size_t ws_size,
                              hipStream_t stream)
{
    const float* x    = (const float*)d_in[0];
    const int*   e0   = (const int*)  d_in[1];
    const int*   e1   = (const int*)  d_in[2];
    const float* Wl   = (const float*)d_in[3];
    const float* bl   = (const float*)d_in[4];
    const float* Wr   = (const float*)d_in[5];
    const float* br   = (const float*)d_in[6];
    const float* attn = (const float*)d_in[7];
    const float* ral  = (const float*)d_in[8];
    const float* rar  = (const float*)d_in[9];

    float* ws = (float*)d_ws;
    const size_t ND = (size_t)NNODES * DDIM;
    const size_t DD = (size_t)DDIM * DDIM;
    float* l     = ws;
    float* r     = ws + ND;
    float* o0    = ws + 2 * ND;
    float* o1    = ws + 3 * ND;
    float* exbuf = ws + 4 * ND;                         // E*H
    float* mx    = exbuf + (size_t)EEDGES * HEADS;      // N*H
    float* den   = mx + (size_t)NNODES * HEADS;         // N*H
    unsigned short* xbf = (unsigned short*)(den + (size_t)NNODES * HEADS);  // N*D bf16
    unsigned short* wbf = xbf + ND;                     // 2*D*D bf16

    // 0) one-shot fp32 -> bf16 conversion of GEMM operands
    const int xn4 = (int)(ND / 4), wn4 = (int)(DD / 4);
    f32_to_bf16_kernel<<<(xn4 + 255) / 256, 256, 0, stream>>>(x, xbf, xn4);
    f32_to_bf16_kernel<<<(wn4 + 255) / 256, 256, 0, stream>>>(Wl, wbf, wn4);
    f32_to_bf16_kernel<<<(wn4 + 255) / 256, 256, 0, stream>>>(Wr, wbf + DD, wn4);

    // 1) fused dual projection GEMM (TDM-staged, z: 0 -> l/W_l, 1 -> r/W_r)
    dim3 gGemm((NNODES + 63) / 64, DDIM / 64, 2);
    proj_gemm_tdm_kernel<<<gGemm, 128, 0, stream>>>(xbf, wbf, bl, br, l, r);

    // 2) per-metapath attention aggregation
    const int*  eis[2]  = {e0, e1};
    float*      outs[2] = {o0, o1};
    for (int m = 0; m < 2; ++m) {
        (void)hipMemsetAsync(outs[m], 0, ND * sizeof(float), stream);
        (void)hipMemsetAsync(den, 0, (size_t)NNODES * HEADS * sizeof(float), stream);
        fill_neginf_kernel<<<(NNODES * HEADS + 255) / 256, 256, 0, stream>>>(
            (unsigned int*)mx, NNODES * HEADS);
        edge_logits_kernel<<<(EEDGES + 7) / 8, 256, 0, stream>>>(
            eis[m], l, r, attn + m * HEADS * 2 * CHAN, exbuf, mx);
        edge_exp_kernel<<<(EEDGES * HEADS + 255) / 256, 256, 0, stream>>>(
            eis[m], exbuf, mx, den);
        edge_scatter_kernel<<<(EEDGES + 7) / 8, 256, 0, stream>>>(
            eis[m], l, exbuf, den, outs[m]);
    }

    // 3) relation-level beta attention + ReLU
    final_kernel<<<(NNODES + 7) / 8, 256, 0, stream>>>(r, o0, o1, ral, rar, (float*)d_out);
}
